// ChiralEmbeddingModel_3513283248582
// MI455X (gfx1250) — compile-verified, matrix operand order checked by reading, not run on hardware
//
#include <hip/hip_runtime.h>
#include <hip/hip_bf16.h>
#include <math.h>

// ---------------------------------------------------------------------------
// Types
// ---------------------------------------------------------------------------
typedef __attribute__((ext_vector_type(16))) __bf16         v16bf;
typedef __attribute__((ext_vector_type(8)))  float          v8f;
typedef __attribute__((ext_vector_type(8)))  unsigned short us8;

#define DEV __device__ __forceinline__

// Problem constants
#define N_C     128
#define N_INV   128
#define N_K     128
#define N_OUT   256
#define N_HID   256
#define EMB_W   512            // INV + 3*C
#define EPS_F   1e-5f

// Scales from the reference
#define S_LIN   0.08838834764831845f   // 1/sqrt(128)
#define S_CROSS 0.0625f                // 1/sqrt(2*128)
#define S_DOT   0.05103103630798288f   // 1/sqrt(3*128)

// ---------------------------------------------------------------------------
// BF16 helpers — use native converts (v_cvt_*bf16_f32) via __bf16 casts
// ---------------------------------------------------------------------------
DEV unsigned short f2bf(float f) {
    union { __bf16 b; unsigned short u; } x;
    x.b = (__bf16)f;
    return x.u;
}
DEV float bf2f(unsigned short h) {
    union { float f; unsigned u; } v; v.u = ((unsigned)h) << 16; return v.f;
}
// Unpack a b32 holding two bf16 (lo = channel u, hi = channel u+1)
DEV float bfl(unsigned p) { union { float f; unsigned u; } v; v.u = p << 16;            return v.f; }
DEV float bfh(unsigned p) { union { float f; unsigned u; } v; v.u = p & 0xFFFF0000u;    return v.f; }
// Pack two floats into two bf16 in one b32
DEV unsigned pk2bf(float a, float b) {
    return (unsigned)f2bf(a) | ((unsigned)f2bf(b) << 16);
}

union BfFrag { us8 s[2]; v16bf v; };

// A-fragment (16x32 bf16, M x K) from a row-major [16][aCols] bf16 LDS buffer.
// ISA layout: lanes 0-15: M=lane, halves 0..7 -> K=0..7, halves 8..15 -> K=16..23;
// lanes 16-31: M=lane-16, K offset +8.  => two 16B LDS loads per lane.
DEV v16bf lds_a_frag(const unsigned short* base, int aCols, int kt, int lane) {
    int m    = lane & 15;
    int koff = kt * 32 + ((lane >> 4) << 3);
    const unsigned short* p = base + m * aCols + koff;
    BfFrag u;
    u.s[0] = *(const us8*)(p);
    u.s[1] = *(const us8*)(p + 16);
    return u.v;
}

// B-fragment (32x16 bf16, K x N) from fragment-major bf16 weights in global
// memory (laid out by prep kernel): frag = (kt*NT + nt), 32 lanes x 16 halves
// contiguous -> two global_load_b128 per lane.
DEV v16bf gbl_b_frag(const unsigned short* wf, int NT, int kt, int nt, int lane) {
    const unsigned short* p = wf + ((((kt * NT) + nt) * 32 + lane) << 4);
    BfFrag u;
    u.s[0] = *(const us8*)(p);
    u.s[1] = *(const us8*)(p + 8);
    return u.v;
}

DEV v8f bwmma(v16bf a, v16bf b, v8f c) {
    // D = A(16x32 bf16) x B(32x16 bf16) + C(16x16 f32)
    return __builtin_amdgcn_wmma_f32_16x16x32_bf16(
        /*neg_a=*/false, a, /*neg_b=*/false, b,
        /*c_mod=*/(short)0, c, /*reuse_a=*/false, /*reuse_b=*/false);
}

// Scatter a 16x16 f32 D tile (C/D layout: VGPR r -> M = r + 8*(lane>>4),
// N = lane&15) into a row-major bf16 LDS buffer.
DEV void st_d_bf16(unsigned short* base, int cols, int nt, int lane, v8f acc, float scale) {
    int n  = nt * 16 + (lane & 15);
    int mb = (lane >> 4) << 3;
    #pragma unroll
    for (int r = 0; r < 8; ++r)
        base[(mb + r) * cols + n] = f2bf(acc[r] * scale);
}

DEV float wave_sum(float v) {
    #pragma unroll
    for (int m = 16; m >= 1; m >>= 1) v += __shfl_xor(v, m, 32);
    return v;
}

DEV v8f gemm_k128(v16bf a0, v16bf a1, v16bf a2, v16bf a3,
                  const unsigned short* w, int NT, int nt, int lane) {
    v8f acc = {0.f,0.f,0.f,0.f,0.f,0.f,0.f,0.f};
    acc = bwmma(a0, gbl_b_frag(w, NT, 0, nt, lane), acc);
    acc = bwmma(a1, gbl_b_frag(w, NT, 1, nt, lane), acc);
    acc = bwmma(a2, gbl_b_frag(w, NT, 2, nt, lane), acc);
    acc = bwmma(a3, gbl_b_frag(w, NT, 3, nt, lane), acc);
    return acc;
}

// ---------------------------------------------------------------------------
// Workspace layout (bf16 halves): fragment-major weights
// ---------------------------------------------------------------------------
#define OFF_L0   0        // W_lin0   128x128  (4 kt x 8 nt x 512)
#define OFF_L1   16384
#define OFF_L2   32768
#define OFF_CT   49152    // w_cross^T 128x128
#define OFF_DT   65536    // w_dot^T   128x128
#define OFF_G1   81920    // gate_W1 128x256 (4 kt x 16 nt)
#define OFF_G2   114688   // gate_W2 256x128 (8 kt x 8 nt)
#define OFF_WO   147456   // W_out   128x256 (4 kt x 16 nt)

// ---------------------------------------------------------------------------
// Prep: f32 weight -> bf16 WMMA B-fragments (optionally transposed source)
// B[k][n] mapping per lane/half mirrors the 16-bit A layout.
// ---------------------------------------------------------------------------
__global__ void prep_weights(const float* __restrict__ src,
                             unsigned short* __restrict__ dst,
                             int Ksz, int Nsz, int transpose) {
    int NT    = Nsz >> 4;
    int total = (Ksz >> 5) * NT * 512;
    int idx   = blockIdx.x * blockDim.x + threadIdx.x;
    if (idx >= total) return;
    int h    = idx & 15;
    int lane = (idx >> 4) & 31;
    int f    = idx >> 9;
    int nt   = f % NT;
    int kt   = f / NT;
    int kk   = (h & 7) + ((h >> 3) << 4) + ((lane >> 4) << 3);
    int k    = kt * 32 + kk;
    int n    = nt * 16 + (lane & 15);
    float v  = transpose ? src[n * Ksz + k] : src[k * Nsz + n];
    dst[idx] = f2bf(v);
}

// ---------------------------------------------------------------------------
// Main kernel: 1 wave32 per 16-atom tile
// ---------------------------------------------------------------------------
__global__ __launch_bounds__(32)
void chiral_main(const float* __restrict__ emb,
                 const float* __restrict__ mean_inv,
                 const float* __restrict__ std_inv,
                 const float* __restrict__ rms_gamma,
                 const float* __restrict__ ln_gamma,
                 const unsigned short* __restrict__ ws,
                 float* __restrict__ out) {
    __shared__ __align__(16) unsigned char smem[57344];

    unsigned short* eqS  = (unsigned short*)(smem + 0);      // 3 x [16][128] bf16, 12KB
    unsigned short* invS = (unsigned short*)(smem + 12288);  // [16][128] bf16, 4KB
    unsigned short* xst  = (unsigned short*)(smem + 16384);  // [16][128] bf16 stage, 4KB
    unsigned short* x0s  = (unsigned short*)(smem + 20480);  // 3 x [16][128] bf16, 12KB
    unsigned short* y1s  = (unsigned short*)(smem + 32768);  // 3 x [16][128] bf16, 12KB
    unsigned short* y2s  = (unsigned short*)(smem + 45056);  // 3 x [16][128] bf16, 12KB
    unsigned short* hS   = eqS;   // overlay: [16][256] bf16 in phase D (eqS dead)
    unsigned short* psnS = x0s;   // overlay: [16][128] bf16 psn (x0s dead)
    float*          rrms = (float*)xst;  // overlay: 16 floats during staging only

    const int lane     = threadIdx.x;
    const int atomBase = blockIdx.x * 16;

    const unsigned short* wL0 = ws + OFF_L0;
    const unsigned short* wL1 = ws + OFF_L1;
    const unsigned short* wL2 = ws + OFF_L2;
    const unsigned short* wCT = ws + OFF_CT;
    const unsigned short* wDT = ws + OFF_DT;
    const unsigned short* wG1 = ws + OFF_G1;
    const unsigned short* wG2 = ws + OFF_G2;
    const unsigned short* wWO = ws + OFF_WO;

    // ---- Stage 0a: per-atom RMS of eq (384 f32 each) ----
    for (int a = 0; a < 16; ++a) {
        const float* row = emb + (size_t)(atomBase + a) * EMB_W + N_INV;
        float s = 0.f;
        for (int t = lane; t < 384; t += 32) { float v = row[t]; s += v * v; }
        s = wave_sum(s);
        float rms = sqrtf(s * (1.0f / 384.0f) + EPS_F);
        if (lane == 0) rrms[a] = 1.0f / rms;
    }

    // ---- Stage 0b: eq -> normalized bf16 A-tiles per xyz component ----
    for (int it = 0; it < 192; ++it) {
        int idx = it * 32 + lane;
        int m = idx / 384, t = idx - m * 384;
        int c = t / 3,     i = t - c * 3;
        float v = emb[(size_t)(atomBase + m) * EMB_W + N_INV + t];
        v = v * rrms[m] * rms_gamma[c];
        eqS[i * 2048 + m * 128 + c] = f2bf(v);
    }

    // ---- Stage 0c: inv -> standardized bf16 A-tile (paired b32 stores) ----
    for (int it = 0; it < 32; ++it) {
        int idx = it * 64 + lane * 2;
        int m = idx >> 7, c = idx & 127;
        const float* row = emb + (size_t)(atomBase + m) * EMB_W;
        float v0 = (row[c]     - mean_inv[c])     / std_inv[c];
        float v1 = (row[c + 1] - mean_inv[c + 1]) / std_inv[c + 1];
        *(unsigned*)(invS + m * 128 + c) = pk2bf(v0, v1);
    }

    // ---- Phase A: 5 GEMMs per xyz component (all K=128, M=16, N=128) ----
    for (int i = 0; i < 3; ++i) {
        const unsigned short* eqA = eqS + i * 2048;
        v16bf a0 = lds_a_frag(eqA, 128, 0, lane);
        v16bf a1 = lds_a_frag(eqA, 128, 1, lane);
        v16bf a2 = lds_a_frag(eqA, 128, 2, lane);
        v16bf a3 = lds_a_frag(eqA, 128, 3, lane);

        for (int nt = 0; nt < 8; ++nt) {   // x0 and x1
            v8f d0 = gemm_k128(a0, a1, a2, a3, wL0, 8, nt, lane);
            st_d_bf16(x0s + i * 2048, 128, nt, lane, d0, S_LIN);
            v8f d1 = gemm_k128(a0, a1, a2, a3, wL1, 8, nt, lane);
            st_d_bf16(xst, 128, nt, lane, d1, S_LIN);
        }
        {   // y1 = x1 @ w_cross^T
            v16bf c0 = lds_a_frag(xst, 128, 0, lane);
            v16bf c1 = lds_a_frag(xst, 128, 1, lane);
            v16bf c2 = lds_a_frag(xst, 128, 2, lane);
            v16bf c3 = lds_a_frag(xst, 128, 3, lane);
            for (int nt = 0; nt < 8; ++nt) {
                v8f d = gemm_k128(c0, c1, c2, c3, wCT, 8, nt, lane);
                st_d_bf16(y1s + i * 2048, 128, nt, lane, d, 1.0f);
            }
        }
        for (int nt = 0; nt < 8; ++nt) {   // x2
            v8f d2 = gemm_k128(a0, a1, a2, a3, wL2, 8, nt, lane);
            st_d_bf16(xst, 128, nt, lane, d2, S_LIN);
        }
        {   // y2 = x2 @ w_dot^T
            v16bf c0 = lds_a_frag(xst, 128, 0, lane);
            v16bf c1 = lds_a_frag(xst, 128, 1, lane);
            v16bf c2 = lds_a_frag(xst, 128, 2, lane);
            v16bf c3 = lds_a_frag(xst, 128, 3, lane);
            for (int nt = 0; nt < 8; ++nt) {
                v8f d = gemm_k128(c0, c1, c2, c3, wDT, 8, nt, lane);
                st_d_bf16(y2s + i * 2048, 128, nt, lane, d, 1.0f);
            }
        }
    }

    // ---- Phase B: ps[n,u] = <cross(x0,y1), y2> ----
    // Two consecutive u-channels per lane -> all LDS reads are b32.
    // psr[2*it+q] holds ps at global index it*64 + lane*2 + q.
    float psr[64];
    #pragma unroll
    for (int it = 0; it < 32; ++it) {
        int idx = it * 64 + lane * 2;
        int m = idx >> 7, u = idx & 127;
        float x0a[3], x0b[3], y1a[3], y1b[3], y2a[3], y2b[3];
        #pragma unroll
        for (int j = 0; j < 3; ++j) {
            unsigned p0 = *(const unsigned*)(x0s + j * 2048 + m * 128 + u);
            unsigned p1 = *(const unsigned*)(y1s + j * 2048 + m * 128 + u);
            unsigned p2 = *(const unsigned*)(y2s + j * 2048 + m * 128 + u);
            x0a[j] = bfl(p0); x0b[j] = bfh(p0);
            y1a[j] = bfl(p1); y1b[j] = bfh(p1);
            y2a[j] = bfl(p2); y2b[j] = bfh(p2);
        }
        float psA = 0.f, psB = 0.f;
        #pragma unroll
        for (int j = 0; j < 3; ++j) {
            int j1 = (j + 1) % 3, j2 = (j + 2) % 3;
            psA += (x0a[j1] * y1a[j2] - x0a[j2] * y1a[j1]) * S_CROSS * y2a[j];
            psB += (x0b[j1] * y1b[j2] - x0b[j2] * y1b[j1]) * S_CROSS * y2b[j];
        }
        psr[2 * it]     = psA * S_DOT;
        psr[2 * it + 1] = psB * S_DOT;
    }

    // ---- Phase C: LayerNorm over u per atom; psn -> bf16 LDS (b32 stores) ----
    // Atom a's 128 values live at psr[4a + 2p + q], u = p*64 + lane*2 + q.
    #pragma unroll
    for (int a = 0; a < 16; ++a) {
        float s0 = psr[4*a] + psr[4*a+1] + psr[4*a+2] + psr[4*a+3];
        float mu = wave_sum(s0) * (1.0f / 128.0f);
        float vs = 0.f;
        #pragma unroll
        for (int q = 0; q < 4; ++q) { float d = psr[4*a+q] - mu; vs += d * d; }
        float rstd = rsqrtf(wave_sum(vs) * (1.0f / 128.0f) + EPS_F);
        #pragma unroll
        for (int p = 0; p < 2; ++p) {
            int u = p * 64 + lane * 2;
            float n0 = (psr[4*a + 2*p]     - mu) * rstd * ln_gamma[u];
            float n1 = (psr[4*a + 2*p + 1] - mu) * rstd * ln_gamma[u + 1];
            *(unsigned*)(psnS + a * 128 + u) = pk2bf(n0, n1);
        }
    }

    // ---- Phase D1: h = silu(inv @ gate_W1)  (M=16, K=128, N=256) ----
    {
        v16bf i0 = lds_a_frag(invS, 128, 0, lane);
        v16bf i1 = lds_a_frag(invS, 128, 1, lane);
        v16bf i2 = lds_a_frag(invS, 128, 2, lane);
        v16bf i3 = lds_a_frag(invS, 128, 3, lane);
        for (int nt = 0; nt < 16; ++nt) {
            v8f acc = gemm_k128(i0, i1, i2, i3, wG1, 16, nt, lane);
            #pragma unroll
            for (int r = 0; r < 8; ++r) {
                float x = acc[r];
                acc[r] = x / (1.0f + __expf(-x));
            }
            st_d_bf16(hS, 256, nt, lane, acc, 1.0f);
        }
    }

    // ---- Phase D2: g = sigmoid(h @ gate_W2); m = g*psn (K=256, N=128) ----
    for (int nt = 0; nt < 8; ++nt) {
        v8f acc = {0.f,0.f,0.f,0.f,0.f,0.f,0.f,0.f};
        #pragma unroll
        for (int kt = 0; kt < 8; ++kt)
            acc = bwmma(lds_a_frag(hS, 256, kt, lane),
                        gbl_b_frag(wG2, 8, kt, nt, lane), acc);
        int n  = nt * 16 + (lane & 15);
        int mb = (lane >> 4) << 3;
        #pragma unroll
        for (int r = 0; r < 8; ++r) {
            float g = 1.0f / (1.0f + __expf(-acc[r]));
            int m = mb + r;
            xst[m * 128 + n] = f2bf(g * bf2f(psnS[m * 128 + n]));
        }
    }

    // ---- Phase D3: out = (g*psn) @ W_out  (M=16, K=128, N=256) ----
    {
        v16bf m0 = lds_a_frag(xst, 128, 0, lane);
        v16bf m1 = lds_a_frag(xst, 128, 1, lane);
        v16bf m2 = lds_a_frag(xst, 128, 2, lane);
        v16bf m3 = lds_a_frag(xst, 128, 3, lane);
        for (int nt = 0; nt < 16; ++nt) {
            v8f acc = gemm_k128(m0, m1, m2, m3, wWO, 16, nt, lane);
            int n  = nt * 16 + (lane & 15);
            int mb = (lane >> 4) << 3;
            #pragma unroll
            for (int r = 0; r < 8; ++r)
                out[(size_t)(atomBase + mb + r) * N_OUT + n] = acc[r];
        }
    }
}

// ---------------------------------------------------------------------------
// Launch
// ---------------------------------------------------------------------------
extern "C" void kernel_launch(void* const* d_in, const int* in_sizes, int n_in,
                              void* d_out, int out_size, void* d_ws, size_t ws_size,
                              hipStream_t stream) {
    const float* emb       = (const float*)d_in[0];
    const float* mean_inv  = (const float*)d_in[1];
    const float* std_inv   = (const float*)d_in[2];
    const float* rms_gamma = (const float*)d_in[3];
    const float* W_lin0    = (const float*)d_in[4];
    const float* W_lin1    = (const float*)d_in[5];
    const float* W_lin2    = (const float*)d_in[6];
    const float* w_cross   = (const float*)d_in[7];
    const float* w_dot     = (const float*)d_in[8];
    const float* ln_gamma  = (const float*)d_in[9];
    const float* gate_W1   = (const float*)d_in[10];
    const float* gate_W2   = (const float*)d_in[11];
    const float* W_out     = (const float*)d_in[12];

    unsigned short* ws = (unsigned short*)d_ws;
    float* out = (float*)d_out;

    auto prep = [&](const float* src, int off, int K, int N, int tr) {
        int total = (K / 32) * (N / 16) * 512;
        prep_weights<<<(total + 255) / 256, 256, 0, stream>>>(src, ws + off, K, N, tr);
    };
    prep(W_lin0,  OFF_L0, 128, 128, 0);
    prep(W_lin1,  OFF_L1, 128, 128, 0);
    prep(W_lin2,  OFF_L2, 128, 128, 0);
    prep(w_cross, OFF_CT, 128, 128, 1);   // B[k=v][n=u] = w_cross[u][v]
    prep(w_dot,   OFF_DT, 128, 128, 1);
    prep(gate_W1, OFF_G1, 128, 256, 0);
    prep(gate_W2, OFF_G2, 256, 128, 0);
    prep(W_out,   OFF_WO, 128, 256, 0);

    int n = in_sizes[0] / EMB_W;      // 131072 atoms
    int nblk = n / 16;                // 16 atoms per wave32 workgroup
    if (nblk < 1) nblk = 1;
    chiral_main<<<nblk, 32, 0, stream>>>(emb, mean_inv, std_inv, rms_gamma,
                                         ln_gamma, ws, out);
}
